// ConditionalGenerator_28664611733735
// MI455X (gfx1250) — matrix operand order repaired and checked by reading, not compiled, not run on hardware
//
#include <hip/hip_runtime.h>

// ---------------- problem constants ----------------
#define HID    64
#define NOI    16
#define MLPW   128
#define NCOND  4
#define NDATA  8
#define TSTEPS 128
#define BATCH  8192
#define ROWS   64                 // batch rows per block
#define NBLK   (BATCH / ROWS)     // 128 blocks
#define DTV    (1.0f / 127.0f)

// ---------------- vector / address-space types ----------------
typedef __attribute__((ext_vector_type(16))) __bf16         v16bf;
typedef __attribute__((ext_vector_type(16))) unsigned short v16u;
typedef __attribute__((ext_vector_type(8)))  unsigned short v8u;
typedef __attribute__((ext_vector_type(8)))  float          v8f;

typedef const __attribute__((address_space(1))) unsigned short* gus_p;
typedef const __attribute__((address_space(1))) float*          gfl_p;
typedef __attribute__((address_space(1))) float*                gfw_p;
typedef const __attribute__((address_space(1))) v16u*           gv16u_p;

static __device__ __forceinline__ gus_p GUS(const unsigned short* p) {
  return (gus_p)(unsigned long long)p;
}
static __device__ __forceinline__ gfl_p GFL(const float* p) {
  return (gfl_p)(unsigned long long)p;
}
static __device__ __forceinline__ gfw_p GFW(float* p) {
  return (gfw_p)(unsigned long long)p;
}

static __device__ __forceinline__ unsigned short f2bf(float f) {
  unsigned int u = __float_as_uint(f);
  u += 0x7FFFu + ((u >> 16) & 1u);           // round-to-nearest-even
  return (unsigned short)(u >> 16);
}
// fast sigmoid/tanh: single v_exp_f32 + v_rcp_f32 (no IEEE divide fixup chain)
static __device__ __forceinline__ float lipswish(float x) {
  return 0.909f * x * __builtin_amdgcn_rcpf(1.0f + __expf(-x));
}
static __device__ __forceinline__ float fast_tanh(float x) {
  float e = __expf(2.0f * x);                // inf for large x -> rcp 0 -> +1
  return 1.0f - 2.0f * __builtin_amdgcn_rcpf(e + 1.0f);
}

// Async global->LDS copy of one dword (ASYNCcnt tracked); lds_off = LDS byte addr
static __device__ __forceinline__ void async_g2l_b32(unsigned lds_off,
                                                     unsigned long long gaddr) {
  asm volatile("global_load_async_to_lds_b32 %0, %1, off"
               :: "v"(lds_off), "v"(gaddr) : "memory");
}
static __device__ __forceinline__ void wait_async0() {
  asm volatile("s_wait_asynccnt 0x0" ::: "memory");
}
// LDS generic addresses carry the byte offset in bits [31:0]
static __device__ __forceinline__ unsigned lds_off32(const void* p) {
  return (unsigned)(unsigned long long)p;
}

// A fragment: 16x32 bf16 tile, row-major LDS. lane = M row, halves = K.
static __device__ __forceinline__ v16bf loadA(const unsigned short* base, int ld,
                                              int r0, int kbase, int lane) {
  int m   = lane & 15;
  int klo = kbase + ((lane < 16) ? 0 : 8);
  int khi = kbase + ((lane < 16) ? 16 : 24);
  v8u lo = *(const v8u*)(base + (r0 + m) * ld + klo);
  v8u hi = *(const v8u*)(base + (r0 + m) * ld + khi);
  v16u r;
#pragma unroll
  for (int i = 0; i < 8; ++i) { r[i] = lo[i]; r[8 + i] = hi[i]; }
  return __builtin_bit_cast(v16bf, r);
}
// B fragment: 32x16 bf16 tile from row-major weights in GLOBAL memory.
static __device__ __forceinline__ v16bf loadB(gus_p base, int ld, int kbase, int n0,
                                              int lane) {
  int n = lane & 15, hk = (lane >> 4) * 8;
  v16u r;
#pragma unroll
  for (int s = 0; s < 16; ++s)
    r[s] = base[(kbase + ((s < 8) ? hk + s : 16 + hk + (s - 8))) * ld + n0 + n];
  return __builtin_bit_cast(v16bf, r);
}
static __device__ __forceinline__ v8f wmma_bf16(v16bf a, v16bf b, v8f c) {
  return __builtin_amdgcn_wmma_f32_16x16x32_bf16(false, a, false, b, (short)0, c,
                                                 false, false);
}

// ---------------- LDS working set (~194 KB / WGP, 320 KB budget) ----------------
struct Shared {
  unsigned short X[ROWS * 64];     //  8 KB  z-state (bf16), layer-1 A operand
  unsigned short H0[ROWS * 256];   // 32 KB  layer-1 out (dr|di), bf16
  unsigned short H1[ROWS * 256];   // 32 KB  layer-2 out (dr|di), bf16
  float Y[ROWS * 64];              // 16 KB  y carry
  float Z[ROWS * 64];              // 16 KB  z carry
  float F[ROWS * 64];              // 16 KB  f carry
  float Fn[ROWS * 64];             // 16 KB  new f
  float G[ROWS * 64];              // 16 KB  g·dW carry (has this step's dW folded)
  float A[ROWS * 64];              // 16 KB  g_{n+1}·dW[n]
  float Gn[ROWS * 64];             // 16 KB  g_{n+1}·dW[n+1]
  float DWc[ROWS * NOI];           //  4 KB  dW[n]   (async-filled)
  float DWn[ROWS * NOI];           //  4 KB  dW[n+1] (async-filled)
  float RoT[NDATA * HID];          //  2 KB  ro_W^T
  float RoC[NDATA];                //        ro bias + cond contribution
};

// Evaluate both MLPs at the state in s.X. Layer-1 bias folds t & c0 (beff_t).
// Diffusion output tiles are contracted in-register against dW[n] / dW[n+1],
// which arrive via async global->LDS issued before this call (waited below).
static __device__ __forceinline__ void eval_mlps(
    Shared& s, gus_p w0c, gus_p w1dr, gus_p w1di, gus_p w2dr, gus_p w2di,
    gfl_p beff_t, gfl_p b1dr, gfl_p b1di, gfl_p b2dr, gfl_p b2di, bool do_cur,
    bool do_next, int tid) {
  const int wave = tid >> 5, lane = tid & 31;
  const int m = lane & 15, rh = (lane >> 4) << 3;

  // ---- Layer 1: X(64x64) @ W0c(64x256) -> H0 (lipswish). B hoisted over rows.
#pragma unroll
  for (int c = 0; c < 2; ++c) {
    int n0 = (wave * 2 + c) * 16;
    v16bf b0 = loadB(w0c, 256, 0, n0, lane);
    v16bf b1 = loadB(w0c, 256, 32, n0, lane);
    int col = n0 + m;
    float bias = beff_t[col];
#pragma unroll
    for (int i = 0; i < 4; ++i) {
      int r0 = i * 16;
      v8f acc = {};
      acc = wmma_bf16(loadA(s.X, 64, r0, 0, lane), b0, acc);
      acc = wmma_bf16(loadA(s.X, 64, r0, 32, lane), b1, acc);
#pragma unroll
      for (int v = 0; v < 8; ++v)
        s.H0[(r0 + v + rh) * 256 + col] = f2bf(lipswish(acc[v] + bias));
    }
  }
  __syncthreads();

  // ---- Layer 2: H0 -> H1 (cols 0..127 drift, 128..255 diffusion)
#pragma unroll
  for (int c = 0; c < 2; ++c) {
    int ct = wave * 2 + c;
    bool dr = ct < 8;
    int n0 = (ct & 7) * 16;
    gus_p wB = dr ? w1dr : w1di;
    const unsigned short* aB = s.H0 + (dr ? 0 : 128);
    v16bf bf[4];
#pragma unroll
    for (int kt = 0; kt < 4; ++kt) bf[kt] = loadB(wB, 128, kt * 32, n0, lane);
    int nl = n0 + m;
    float bias = dr ? b1dr[nl] : b1di[nl];
    int colOut = (dr ? 0 : 128) + nl;
#pragma unroll
    for (int i = 0; i < 4; ++i) {
      int r0 = i * 16;
      v8f acc = {};
#pragma unroll
      for (int kt = 0; kt < 4; ++kt)
        acc = wmma_bf16(loadA(aB, 256, r0, kt * 32, lane), bf[kt], acc);
#pragma unroll
      for (int v = 0; v < 8; ++v)
        s.H1[(r0 + v + rh) * 256 + colOut] = f2bf(lipswish(acc[v] + bias));
    }
  }
  // dW async copies (issued by this wave before eval) must land before the
  // barrier so every wave sees them in the diffusion phase below.
  wait_async0();
  __syncthreads();

  // ---- Layer 3 drift: H1[:,0:128] @ W2dr(128x64) -> tanh -> Fn (waves 0..3)
  if (wave < 4) {
    int n0 = wave * 16;
    v16bf bf[4];
#pragma unroll
    for (int kt = 0; kt < 4; ++kt) bf[kt] = loadB(w2dr, 64, kt * 32, n0, lane);
    float bias = b2dr[n0 + m];
#pragma unroll
    for (int i = 0; i < 4; ++i) {
      int r0 = i * 16;
      v8f acc = {};
#pragma unroll
      for (int kt = 0; kt < 4; ++kt)
        acc = wmma_bf16(loadA(s.H1, 256, r0, kt * 32, lane), bf[kt], acc);
#pragma unroll
      for (int v = 0; v < 8; ++v)
        s.Fn[(r0 + v + rh) * 64 + n0 + m] = fast_tanh(acc[v] + bias);
    }
  }

  // ---- Layer 3 diffusion: H1[:,128:256] @ W2di(128x1024).
  // Cache the whole A-panel in registers (16 frags = 128 VGPRs; fine at 1 WG/WGP,
  // gfx1250 VGPR-MSB addressing) so W2di is fetched exactly once per block-step.
  v16bf aD[16];
#pragma unroll
  for (int i = 0; i < 4; ++i)
#pragma unroll
    for (int kt = 0; kt < 4; ++kt)
      aD[i * 4 + kt] = loadA(s.H1 + 128, 256, i * 16, kt * 32, lane);

  for (int hh = 0; hh < 8; ++hh) {
    int h = wave * 8 + hh;
    int n0 = h * 16;
    if (hh + 1 < 8)  // prefetch next streamed B tile
      __builtin_prefetch((const unsigned short*)w2di + lane * 1024 + n0 + 16, 0, 0);
    v16bf bf[4];
#pragma unroll
    for (int kt = 0; kt < 4; ++kt) bf[kt] = loadB(w2di, 1024, kt * 32, n0, lane);
    float bias = b2di[n0 + m];
#pragma unroll
    for (int i = 0; i < 4; ++i) {
      int r0 = i * 16;
      v8f acc = {};
#pragma unroll
      for (int kt = 0; kt < 4; ++kt) acc = wmma_bf16(aD[i * 4 + kt], bf[kt], acc);
#pragma unroll
      for (int v = 0; v < 8; ++v) {
        int row = r0 + v + rh;
        float g = fast_tanh(acc[v] + bias);  // 16 cols of tile h == noise dim m
        float pc = do_cur ? g * s.DWc[row * NOI + m] : 0.0f;
        float pn = do_next ? g * s.DWn[row * NOI + m] : 0.0f;
#pragma unroll
        for (int off = 1; off < 16; off <<= 1) {  // reduce over the 16 m-lanes
          pc += __shfl_xor(pc, off, 32);
          pn += __shfl_xor(pn, off, 32);
        }
        if (m == 0) {
          if (do_cur)  s.A[row * 64 + h]  = pc;
          if (do_next) s.Gn[row * 64 + h] = pn;
        }
      }
    }
  }
  __syncthreads();
}

// ---------------- main persistent kernel: whole scan per block ----------------
__global__ __launch_bounds__(256) void sde_scan_kernel(
    const float* ts_, const float* dW_, const unsigned short* w0c_,
    const unsigned short* w1dr_, const unsigned short* w1di_,
    const unsigned short* w2dr_, const unsigned short* w2di_, const float* beff_,
    const float* x0_, const float* roT_, const float* roC_, const float* b1dr_,
    const float* b1di_, const float* b2dr_, const float* b2di_, float* out_) {
  __shared__ Shared s;
  const int tid = threadIdx.x;
  const int gr0 = blockIdx.x * ROWS;

  gfl_p ts = GFL(ts_), dW = GFL(dW_), beff = GFL(beff_), x0 = GFL(x0_);
  gfl_p b1dr = GFL(b1dr_), b1di = GFL(b1di_), b2dr = GFL(b2dr_), b2di = GFL(b2di_);
  gus_p w0c = GUS(w0c_), w1dr = GUS(w1dr_), w1di = GUS(w1di_);
  gus_p w2dr = GUS(w2dr_), w2di = GUS(w2di_);
  gfw_p out = GFW(out_);

  // init: y = z = x0 (x0 identical for all batch rows); cache readout mats in LDS
  for (int e = tid; e < ROWS * 64; e += 256) {
    float v = x0[e & 63];
    s.Y[e] = v; s.Z[e] = v; s.X[e] = f2bf(v);
  }
  for (int e = tid; e < NDATA * HID; e += 256) s.RoT[e] = GFL(roT_)[e];
  if (tid < NDATA) s.RoC[tid] = GFL(roC_)[tid];
  // async-stage dW[0] into DWn for the initial diffusion contraction
  for (int e = tid; e < ROWS * NOI; e += 256)
    async_g2l_b32(lds_off32(&s.DWn[e]),
                  (unsigned long long)(dW_ +
                      (size_t)(gr0 + (e >> 4)) * NOI + (e & 15)));
  __syncthreads();

  // f0 and g0·dW[0]
  eval_mlps(s, w0c, w1dr, w1di, w2dr, w2di, beff, b1dr, b1di, b2dr, b2di,
            /*do_cur=*/false, /*do_next=*/true, tid);
  for (int e = tid; e < ROWS * 64; e += 256) { s.F[e] = s.Fn[e]; s.G[e] = s.Gn[e]; }
  __syncthreads();

  // out[:, 0] = readout(x0)
  for (int idx = tid; idx < ROWS * NDATA; idx += 256) {
    int row = idx >> 3, d = idx & 7;
    float acc = s.RoC[d];
#pragma unroll 8
    for (int h = 0; h < 64; ++h) acc += s.Y[row * 64 + h] * s.RoT[d * 64 + h];
    size_t o = ((size_t)(gr0 + row) * TSTEPS) * (1 + NDATA);
    out[o + 1 + d] = acc;
    if (d == 0) out[o] = ts[0];
  }
  __syncthreads();

  for (int n = 0; n < TSTEPS - 1; ++n) {
    bool hasNext = (n + 1) < (TSTEPS - 1);
    // kick off the dW[n] / dW[n+1] tiles async; they land before the diffusion
    // phase (waited inside eval_mlps), overlapping with the L1/L2 GEMMs.
    for (int e = tid; e < ROWS * NOI; e += 256) {
      const float* gc = dW_ + ((size_t)n * BATCH + (gr0 + (e >> 4))) * NOI + (e & 15);
      async_g2l_b32(lds_off32(&s.DWc[e]), (unsigned long long)gc);
      if (hasNext)
        async_g2l_b32(lds_off32(&s.DWn[e]),
                      (unsigned long long)(gc + (size_t)BATCH * NOI));
    }
    // z_{n+1} = 2y - z + f*dt + g·dW[n]  (G already holds the contraction)
    for (int e = tid; e < ROWS * 64; e += 256) {
      float zn = 2.0f * s.Y[e] - s.Z[e] + s.F[e] * DTV + s.G[e];
      s.Z[e] = zn; s.X[e] = f2bf(zn);
    }
    __syncthreads();

    eval_mlps(s, w0c, w1dr, w1di, w2dr, w2di, beff + (size_t)(n + 1) * 256, b1dr,
              b1di, b2dr, b2di, /*do_cur=*/true, /*do_next=*/hasNext, tid);

    // y_{n+1} = y + dt/2 (f + f1) + 1/2 (g·dW[n] + g1·dW[n]); roll carry
    for (int e = tid; e < ROWS * 64; e += 256) {
      float yn = s.Y[e] + 0.5f * DTV * (s.F[e] + s.Fn[e]) + 0.5f * (s.G[e] + s.A[e]);
      s.Y[e] = yn;
      s.F[e] = s.Fn[e];
      if (hasNext) s.G[e] = s.Gn[e];
    }
    __syncthreads();

    // fused readout: out[:, n+1] = [y, c] @ ro_W + ro_b
    for (int idx = tid; idx < ROWS * NDATA; idx += 256) {
      int row = idx >> 3, d = idx & 7;
      float acc = s.RoC[d];
#pragma unroll 8
      for (int h = 0; h < 64; ++h) acc += s.Y[row * 64 + h] * s.RoT[d * 64 + h];
      size_t o = ((size_t)(gr0 + row) * TSTEPS + (n + 1)) * (1 + NDATA);
      out[o + 1 + d] = acc;
      if (d == 0) out[o] = ts[n + 1];
    }
    __syncthreads();
  }
}

// ---------------- prep kernels (run every call; deterministic) ----------------
__global__ void prep_w0c(const float* drW0, const float* diW0, unsigned short* w0c) {
  int idx = blockIdx.x * 256 + threadIdx.x;
  if (idx >= 64 * 256) return;
  int r = idx >> 8, j = idx & 255;  // skip t row (0) and c rows (65..68)
  float v = (j < 128) ? drW0[(1 + r) * 128 + j] : diW0[(1 + r) * 128 + (j - 128)];
  w0c[r * 256 + j] = f2bf(v);
}
__global__ void prep_cast(const float* src, unsigned short* dst, int n) {
  int idx = blockIdx.x * 256 + threadIdx.x;
  if (idx < n) dst[idx] = f2bf(src[idx]);
}
__global__ void prep_beff(const float* ts, const float* cond, const float* drW0,
                          const float* drB0, const float* diW0, const float* diB0,
                          float* beff) {
  int t = blockIdx.x, j = threadIdx.x;
  float tv = ts[t], v;
  if (j < 128) {
    v = drB0[j] + tv * drW0[j];
    for (int c = 0; c < NCOND; ++c) v += cond[c] * drW0[(65 + c) * 128 + j];
  } else {
    int j2 = j - 128;
    v = diB0[j2] + tv * diW0[j2];
    for (int c = 0; c < NCOND; ++c) v += cond[c] * diW0[(65 + c) * 128 + j2];
  }
  beff[t * 256 + j] = v;
}
__global__ void prep_misc(const float* cond, const float* initW, const float* initB,
                          const float* roW, const float* roB, float* x0, float* roT,
                          float* roC) {
  int tid = threadIdx.x;
  if (tid < HID) {  // x0 = ones-rows sum + cond rows + bias (batch-uniform)
    float v = initB[tid];
    for (int r = 0; r < 16; ++r) v += initW[r * HID + tid];
    for (int c = 0; c < NCOND; ++c) v += cond[c] * initW[(16 + c) * HID + tid];
    x0[tid] = v;
  }
  for (int e = tid; e < NDATA * HID; e += 256) {  // ro_W transposed
    int d = e >> 6, h = e & 63;
    roT[d * HID + h] = roW[h * NDATA + d];
  }
  if (tid < NDATA) {
    float v = roB[tid];
    for (int c = 0; c < NCOND; ++c) v += cond[c] * roW[(HID + c) * NDATA + tid];
    roC[tid] = v;
  }
}

extern "C" void kernel_launch(void* const* d_in, const int* in_sizes, int n_in,
                              void* d_out, int out_size, void* d_ws, size_t ws_size,
                              hipStream_t stream) {
  (void)in_sizes; (void)n_in; (void)out_size; (void)ws_size;
  const float* ts    = (const float*)d_in[0];
  const float* cond  = (const float*)d_in[1];
  const float* dW    = (const float*)d_in[2];
  const float* drW0  = (const float*)d_in[3];
  const float* drB0  = (const float*)d_in[4];
  const float* drW1  = (const float*)d_in[5];
  const float* drB1  = (const float*)d_in[6];
  const float* drW2  = (const float*)d_in[7];
  const float* drB2  = (const float*)d_in[8];
  const float* diW0  = (const float*)d_in[9];
  const float* diB0  = (const float*)d_in[10];
  const float* diW1  = (const float*)d_in[11];
  const float* diB1  = (const float*)d_in[12];
  const float* diW2  = (const float*)d_in[13];
  const float* diB2  = (const float*)d_in[14];
  const float* initW = (const float*)d_in[15];
  const float* initB = (const float*)d_in[16];
  const float* roW   = (const float*)d_in[17];
  const float* roB   = (const float*)d_in[18];
  float* out = (float*)d_out;

  // scratch carving (bytes)
  unsigned char* w = (unsigned char*)d_ws;
  unsigned short* w0c  = (unsigned short*)(w + 0);        // 64*256 bf16
  unsigned short* w1dr = (unsigned short*)(w + 32768);    // 128*128 bf16
  unsigned short* w1di = (unsigned short*)(w + 65536);    // 128*128 bf16
  unsigned short* w2dr = (unsigned short*)(w + 98304);    // 128*64 bf16
  unsigned short* w2di = (unsigned short*)(w + 114688);   // 128*1024 bf16
  float* beff = (float*)(w + 376832);                     // 128*256 f32
  float* x0   = (float*)(w + 507904);                     // 64 f32
  float* roT  = (float*)(w + 508160);                     // 8*64 f32
  float* roC  = (float*)(w + 510208);                     // 8 f32

  prep_w0c<<<64, 256, 0, stream>>>(drW0, diW0, w0c);
  prep_cast<<<64, 256, 0, stream>>>(drW1, w1dr, 128 * 128);
  prep_cast<<<64, 256, 0, stream>>>(diW1, w1di, 128 * 128);
  prep_cast<<<32, 256, 0, stream>>>(drW2, w2dr, 128 * 64);
  prep_cast<<<512, 256, 0, stream>>>(diW2, w2di, 128 * 1024);
  prep_beff<<<TSTEPS, 256, 0, stream>>>(ts, cond, drW0, drB0, diW0, diB0, beff);
  prep_misc<<<1, 256, 0, stream>>>(cond, initW, initB, roW, roB, x0, roT, roC);

  sde_scan_kernel<<<NBLK, 256, 0, stream>>>(ts, dW, w0c, w1dr, w1di, w2dr, w2di,
                                            beff, x0, roT, roC, drB1, diB1, drB2,
                                            diB2, out);
}
